// IGMGM_50792283242619
// MI455X (gfx1250) — matrix operand order repaired and checked by reading, not compiled, not run on hardware
//
#include <hip/hip_runtime.h>
#include <math.h>

// ---------------------------------------------------------------------------
// CDNA5 (gfx1250) bf16 WMMA GEMM pipeline for the IGMGM graph-matching loop.
// v2: 2x2 register blocking (32x32 per wave, 4 accumulators -> each fragment
// feeds two v_wmma_f32_16x16x32_bf16) + packed b32 stores for transposed bf16.
// ---------------------------------------------------------------------------

typedef __bf16 bf16_t;
typedef __attribute__((ext_vector_type(16))) __bf16 v16bf;
typedef __attribute__((ext_vector_type(8)))  __bf16 bf16x8;
typedef __attribute__((ext_vector_type(8)))  float  v8f;

#define G_CNT   16
#define NN      128      // nodes per graph == universe size
#define NN2     (NN*NN)  // 16384
#define BIGN    2048     // G_CNT*NN

__device__ __forceinline__ bf16_t f2bf(float f) {
  unsigned u = __builtin_bit_cast(unsigned, f);
  u += 0x7FFFu + ((u >> 16) & 1u);               // round-to-nearest-even
  unsigned short h = (unsigned short)(u >> 16);
  return __builtin_bit_cast(bf16_t, h);
}

__device__ __forceinline__ unsigned pack2bf(float a, float b) {
  unsigned ua = __builtin_bit_cast(unsigned, a);
  ua += 0x7FFFu + ((ua >> 16) & 1u);
  unsigned ub = __builtin_bit_cast(unsigned, b);
  ub += 0x7FFFu + ((ub >> 16) & 1u);
  return (ua >> 16) | (ub & 0xFFFF0000u);
}

// --- WMMA fragment helpers (CDNA5 ISA 7.12.2 16-bit layouts) ----------------
// A fragment: 16x32 tile of a row-major bf16 matrix (ld elems/row).
// lane<16 holds row=lane, K groups {0..7, 16..23}; lane>=16 holds {8..15, 24..31}.
__device__ __forceinline__ v16bf load_a_frag(const bf16_t* __restrict__ base,
                                             int ld, int row0, int k0, int lane) {
  int r = lane & 15;
  int g = (lane < 16) ? 0 : 8;
  const bf16_t* p = base + (size_t)(row0 + r) * ld + (k0 + g);
  union { v16bf v; bf16x8 h[2]; } u;
  u.h[0] = *(const bf16x8*)p;          // K = g .. g+7      (16B load)
  u.h[1] = *(const bf16x8*)(p + 16);   // K = g+16 .. g+23  (16B load)
  return u.v;
}

// B fragment: 32x16 tile, sourced from B^T stored row-major (bt[n][k], ld elems/row).
// lane N = lane&15, K groups {0..7, 16..23} (lane<16) / {8..15, 24..31} (lane>=16).
__device__ __forceinline__ v16bf load_b_frag(const bf16_t* __restrict__ bt,
                                             int ld, int k0, int col0, int lane) {
  int n  = lane & 15;
  int kb = k0 + ((lane < 16) ? 0 : 8);
  const bf16_t* p = bt + (size_t)(col0 + n) * ld + kb;
  union { v16bf v; bf16x8 h[2]; } u;
  u.h[0] = *(const bf16x8*)p;
  u.h[1] = *(const bf16x8*)(p + 16);
  return u.v;
}

__device__ __forceinline__ v8f wmma_bf16(v16bf a, v16bf b, v8f c) {
  return __builtin_amdgcn_wmma_f32_16x16x32_bf16(false, a, false, b,
                                                 (short)0, c, false, false);
}

// C/D tile stores. Accumulator element v maps to (m0 + (lane<16?0:8) + v,
// n0 + (lane&15)) per the 32-bit 16x16 C/D layout.
__device__ __forceinline__ void store_f32(float* __restrict__ dst, int ld,
                                          int m0, int n0, int lane, v8f acc) {
  int n = lane & 15, mb = m0 + ((lane < 16) ? 0 : 8);
  #pragma unroll
  for (int v = 0; v < 8; ++v)
    dst[(size_t)(mb + v) * ld + n0 + n] = acc[v];
}

__device__ __forceinline__ void store_bf16_rm(bf16_t* __restrict__ dst, int ld,
                                              int m0, int n0, int lane, v8f acc) {
  int n = lane & 15, mb = m0 + ((lane < 16) ? 0 : 8);
  #pragma unroll
  for (int v = 0; v < 8; ++v)
    dst[(size_t)(mb + v) * ld + n0 + n] = f2bf(acc[v]);
}

// Transposed store: dst[col][row]; consecutive v are contiguous -> packed b32.
__device__ __forceinline__ void store_bf16_tr(bf16_t* __restrict__ dst, int ld,
                                              int m0, int n0, int lane, v8f acc) {
  int n = lane & 15, mb = m0 + ((lane < 16) ? 0 : 8);
  unsigned* p = (unsigned*)(dst + (size_t)(n0 + n) * ld + mb);  // mb even -> 4B aligned
  #pragma unroll
  for (int v = 0; v < 8; v += 2)
    p[v >> 1] = pack2bf(acc[v], acc[v + 1]);
}

__device__ __forceinline__ float wave_sum(float v) {
  #pragma unroll
  for (int off = 16; off > 0; off >>= 1) v += __shfl_xor(v, off, 32);
  return v;
}

// ---------------------------------------------------------------------------
// Prep kernels
// ---------------------------------------------------------------------------
__global__ void k_prep_w(const float* __restrict__ W, bf16_t* __restrict__ Wbf) {
  size_t idx = (size_t)blockIdx.x * blockDim.x + threadIdx.x;
  if (idx < (size_t)BIGN * BIGN) Wbf[idx] = f2bf(W[idx]);
}

__global__ void k_prep_a(const float* __restrict__ A, bf16_t* __restrict__ Abf) {
  int idx = blockIdx.x * blockDim.x + threadIdx.x;   // G*NN*NN
  if (idx >= G_CNT * NN2) return;
  int g = idx >> 14, rem = idx & (NN2 - 1);
  int n = rem >> 7, m = rem & 127;
  Abf[idx] = f2bf(A[(size_t)(g * NN + n) * BIGN + g * NN + m]);
}

__global__ void k_prep_u(const float* __restrict__ U0, float* __restrict__ Uf,
                         bf16_t* __restrict__ Ubf, bf16_t* __restrict__ UTbf) {
  int idx = blockIdx.x * blockDim.x + threadIdx.x;   // G*NN*NN
  if (idx >= G_CNT * NN2) return;
  float x = U0[idx];
  Uf[idx] = x;
  bf16_t h = f2bf(x);
  int g = idx >> 14, rem = idx & (NN2 - 1);
  int n = rem >> 7, u = rem & 127;
  Ubf[idx] = h;
  UTbf[(size_t)g * NN2 + u * NN + n] = h;
}

__global__ void k_init_mask(float* __restrict__ Mmask) {
  int idx = threadIdx.x;                  // 256
  int i = idx >> 4, j = idx & 15;
  Mmask[idx] = (i != j) ? 1.0f : 0.0f;
}

__global__ void k_zero2(float* __restrict__ a, float* __restrict__ b) {
  int idx = threadIdx.x;                  // 256
  a[idx] = 0.0f; b[idx] = 0.0f;
}

// ---------------------------------------------------------------------------
// forward_backup inner-iteration kernels (2x2 register-blocked, 32x32/wave)
// ---------------------------------------------------------------------------

// AU[g] = A_blk[g] @ U[g]   -> AUbf (row-major) + AUTbf (transposed, packed)
__global__ void __launch_bounds__(128) k_au(const bf16_t* __restrict__ Abf,
                                            const bf16_t* __restrict__ UTbf,
                                            bf16_t* __restrict__ AUbf,
                                            bf16_t* __restrict__ AUTbf) {
  int wave = (blockIdx.x * blockDim.x + threadIdx.x) >> 5;
  int lane = threadIdx.x & 31;
  int g = wave >> 4, q = wave & 15;
  int m0 = (q >> 2) << 5, n0 = (q & 3) << 5;
  const bf16_t* A  = Abf  + (size_t)g * NN2;
  const bf16_t* UT = UTbf + (size_t)g * NN2;
  v8f a00 = {}, a01 = {}, a10 = {}, a11 = {};
  #pragma unroll
  for (int k0 = 0; k0 < NN; k0 += 32) {
    v16bf fa0 = load_a_frag(A, NN, m0,      k0, lane);
    v16bf fa1 = load_a_frag(A, NN, m0 + 16, k0, lane);
    v16bf fb0 = load_b_frag(UT, NN, k0, n0,      lane);
    v16bf fb1 = load_b_frag(UT, NN, k0, n0 + 16, lane);
    a00 = wmma_bf16(fa0, fb0, a00);
    a01 = wmma_bf16(fa0, fb1, a01);
    a10 = wmma_bf16(fa1, fb0, a10);
    a11 = wmma_bf16(fa1, fb1, a11);
  }
  bf16_t* AUg  = AUbf  + (size_t)g * NN2;
  bf16_t* AUTg = AUTbf + (size_t)g * NN2;
  store_bf16_rm(AUg, NN, m0,      n0,      lane, a00);
  store_bf16_rm(AUg, NN, m0,      n0 + 16, lane, a01);
  store_bf16_rm(AUg, NN, m0 + 16, n0,      lane, a10);
  store_bf16_rm(AUg, NN, m0 + 16, n0 + 16, lane, a11);
  store_bf16_tr(AUTg, NN, m0,      n0,      lane, a00);
  store_bf16_tr(AUTg, NN, m0,      n0 + 16, lane, a01);
  store_bf16_tr(AUTg, NN, m0 + 16, n0,      lane, a10);
  store_bf16_tr(AUTg, NN, m0 + 16, n0 + 16, lane, a11);
}

// S[g] = U[g]^T @ AU[g]   (f32)
__global__ void __launch_bounds__(128) k_s(const bf16_t* __restrict__ UTbf,
                                           const bf16_t* __restrict__ AUTbf,
                                           float* __restrict__ Sf) {
  int wave = (blockIdx.x * blockDim.x + threadIdx.x) >> 5;
  int lane = threadIdx.x & 31;
  int g = wave >> 4, q = wave & 15;
  int u0 = (q >> 2) << 5, v0 = (q & 3) << 5;
  const bf16_t* UT  = UTbf  + (size_t)g * NN2;
  const bf16_t* AUT = AUTbf + (size_t)g * NN2;
  v8f a00 = {}, a01 = {}, a10 = {}, a11 = {};
  #pragma unroll
  for (int k0 = 0; k0 < NN; k0 += 32) {
    v16bf fa0 = load_a_frag(UT, NN, u0,      k0, lane);
    v16bf fa1 = load_a_frag(UT, NN, u0 + 16, k0, lane);
    v16bf fb0 = load_b_frag(AUT, NN, k0, v0,      lane);
    v16bf fb1 = load_b_frag(AUT, NN, k0, v0 + 16, lane);
    a00 = wmma_bf16(fa0, fb0, a00);
    a01 = wmma_bf16(fa0, fb1, a01);
    a10 = wmma_bf16(fa1, fb0, a10);
    a11 = wmma_bf16(fa1, fb1, a11);
  }
  float* Sg = Sf + (size_t)g * NN2;
  store_f32(Sg, NN, u0,      v0,      lane, a00);
  store_f32(Sg, NN, u0,      v0 + 16, lane, a01);
  store_f32(Sg, NN, u0 + 16, v0,      lane, a10);
  store_f32(Sg, NN, u0 + 16, v0 + 16, lane, a11);
}

// SsumT[i][v][u] = sum_j M_ij S[j][u][v]   (bf16, stored transposed)
__global__ void k_ssum(const float* __restrict__ Sf, const float* __restrict__ Mmask,
                       bf16_t* __restrict__ SsumT) {
  int idx = blockIdx.x * blockDim.x + threadIdx.x;   // G*NN*NN
  if (idx >= G_CNT * NN2) return;
  int i = idx >> 14, rem = idx & (NN2 - 1);
  int u = rem >> 7, v = rem & 127;
  float acc = 0.0f;
  #pragma unroll
  for (int j = 0; j < G_CNT; ++j)
    acc += Mmask[i * G_CNT + j] * Sf[(size_t)j * NN2 + rem];
  SsumT[(size_t)i * NN2 + v * NN + u] = f2bf(acc);
}

// V[i] = sum_j M_ij W4[i,j] @ U[j]  +  AU[i] @ Ssum[i]   (f32)
__global__ void __launch_bounds__(128) k_v(const bf16_t* __restrict__ Wbf,
                                           const bf16_t* __restrict__ UTbf,
                                           const bf16_t* __restrict__ AUbf,
                                           const bf16_t* __restrict__ SsumT,
                                           const float* __restrict__ Mmask,
                                           float* __restrict__ Vf) {
  int wave = (blockIdx.x * blockDim.x + threadIdx.x) >> 5;
  int lane = threadIdx.x & 31;
  int i = wave >> 4, q = wave & 15;
  int m0 = (q >> 2) << 5, n0 = (q & 3) << 5;
  v8f a00 = {}, a01 = {}, a10 = {}, a11 = {};
  for (int j = 0; j < G_CNT; ++j) {
    if (Mmask[i * G_CNT + j] != 0.0f) {                // mask entries are 0/1
      const bf16_t* Wblk = Wbf + (size_t)(i * NN) * BIGN + j * NN;
      const bf16_t* UTj  = UTbf + (size_t)j * NN2;
      #pragma unroll
      for (int k0 = 0; k0 < NN; k0 += 32) {
        v16bf fa0 = load_a_frag(Wblk, BIGN, m0,      k0, lane);
        v16bf fa1 = load_a_frag(Wblk, BIGN, m0 + 16, k0, lane);
        v16bf fb0 = load_b_frag(UTj, NN, k0, n0,      lane);
        v16bf fb1 = load_b_frag(UTj, NN, k0, n0 + 16, lane);
        a00 = wmma_bf16(fa0, fb0, a00);
        a01 = wmma_bf16(fa0, fb1, a01);
        a10 = wmma_bf16(fa1, fb0, a10);
        a11 = wmma_bf16(fa1, fb1, a11);
      }
    }
  }
  const bf16_t* AUi = AUbf  + (size_t)i * NN2;
  const bf16_t* STi = SsumT + (size_t)i * NN2;
  #pragma unroll
  for (int k0 = 0; k0 < NN; k0 += 32) {
    v16bf fa0 = load_a_frag(AUi, NN, m0,      k0, lane);
    v16bf fa1 = load_a_frag(AUi, NN, m0 + 16, k0, lane);
    v16bf fb0 = load_b_frag(STi, NN, k0, n0,      lane);
    v16bf fb1 = load_b_frag(STi, NN, k0, n0 + 16, lane);
    a00 = wmma_bf16(fa0, fb0, a00);
    a01 = wmma_bf16(fa0, fb1, a01);
    a10 = wmma_bf16(fa1, fb0, a10);
    a11 = wmma_bf16(fa1, fb1, a11);
  }
  float* Vi = Vf + (size_t)i * NN2;
  store_f32(Vi, NN, m0,      n0,      lane, a00);
  store_f32(Vi, NN, m0,      n0 + 16, lane, a01);
  store_f32(Vi, NN, m0 + 16, n0,      lane, a10);
  store_f32(Vi, NN, m0 + 16, n0 + 16, lane, a11);
}

// log-domain Sinkhorn: one workgroup per graph, 128x128 tile in LDS.
// Row pass wave-cooperative (stride-1); col pass thread-per-column (stride-128
// over 64 banks -> lanes hit distinct banks). Fuses exp + U/Ubf/UTbf write-out.
__global__ void __launch_bounds__(256) k_sinkhorn(const float* __restrict__ Vf, float tau,
                                                  float* __restrict__ Uf,
                                                  bf16_t* __restrict__ Ubf,
                                                  bf16_t* __restrict__ UTbf) {
  __shared__ float ls[NN2];          // 64 KB
  int g = blockIdx.x;
  const float* V = Vf + (size_t)g * NN2;
  float inv_tau = 1.0f / tau;
  for (int e = threadIdx.x; e < NN2; e += 256) ls[e] = V[e] * inv_tau;
  __syncthreads();

  int lane = threadIdx.x & 31;
  int wid  = threadIdx.x >> 5;       // 8 waves
  for (int it = 0; it < 10; ++it) {
    // rows
    for (int r = wid; r < NN; r += 8) {
      float* row = ls + r * NN;
      float mx = -3.402823e38f;
      #pragma unroll
      for (int c = lane; c < NN; c += 32) mx = fmaxf(mx, row[c]);
      #pragma unroll
      for (int off = 16; off > 0; off >>= 1) mx = fmaxf(mx, __shfl_xor(mx, off, 32));
      float s = 0.0f;
      #pragma unroll
      for (int c = lane; c < NN; c += 32) s += __expf(row[c] - mx);
      s = wave_sum(s);
      float lse = mx + __logf(s);
      #pragma unroll
      for (int c = lane; c < NN; c += 32) row[c] -= lse;
    }
    __syncthreads();
    // cols
    if (threadIdx.x < NN) {
      int c = threadIdx.x;
      float mx = -3.402823e38f;
      for (int r = 0; r < NN; ++r) mx = fmaxf(mx, ls[r * NN + c]);
      float s = 0.0f;
      for (int r = 0; r < NN; ++r) s += __expf(ls[r * NN + c] - mx);
      float lse = mx + __logf(s);
      for (int r = 0; r < NN; ++r) ls[r * NN + c] -= lse;
    }
    __syncthreads();
  }

  for (int e = threadIdx.x; e < NN2; e += 256) {
    float u = __expf(ls[e]);
    Uf[(size_t)g * NN2 + e] = u;
    bf16_t h = f2bf(u);
    int n = e >> 7, uu = e & 127;
    Ubf [(size_t)g * NN2 + e] = h;
    UTbf[(size_t)g * NN2 + uu * NN + n] = h;
  }
}

// ---------------------------------------------------------------------------
// get_alpha kernels (pairwise, off-diagonal only; 2x2 register-blocked)
// ---------------------------------------------------------------------------

// X_ij = U_i @ U_j^T ; store XT (packed); fused term_w = <W4[i,j], X_ij>
__global__ void __launch_bounds__(128) k_x(const bf16_t* __restrict__ Ubf,
                                           const float* __restrict__ W,
                                           bf16_t* __restrict__ XTbf,
                                           float* __restrict__ termw) {
  int wave = (blockIdx.x * blockDim.x + threadIdx.x) >> 5;
  int lane = threadIdx.x & 31;
  int pair = wave >> 4, q = wave & 15;
  int i = pair >> 4, j = pair & 15;
  if (i == j) return;
  int m0 = (q >> 2) << 5, n0 = (q & 3) << 5;   // m0: n-dim rows, n0: m-dim cols
  const bf16_t* Ui = Ubf + (size_t)i * NN2;
  const bf16_t* Uj = Ubf + (size_t)j * NN2;    // serves as B^T (bt[m][u])
  v8f a00 = {}, a01 = {}, a10 = {}, a11 = {};
  #pragma unroll
  for (int k0 = 0; k0 < NN; k0 += 32) {
    v16bf fa0 = load_a_frag(Ui, NN, m0,      k0, lane);
    v16bf fa1 = load_a_frag(Ui, NN, m0 + 16, k0, lane);
    v16bf fb0 = load_b_frag(Uj, NN, k0, n0,      lane);
    v16bf fb1 = load_b_frag(Uj, NN, k0, n0 + 16, lane);
    a00 = wmma_bf16(fa0, fb0, a00);
    a01 = wmma_bf16(fa0, fb1, a01);
    a10 = wmma_bf16(fa1, fb0, a10);
    a11 = wmma_bf16(fa1, fb1, a11);
  }
  bf16_t* XT = XTbf + (size_t)pair * NN2;
  store_bf16_tr(XT, NN, m0,      n0,      lane, a00);
  store_bf16_tr(XT, NN, m0,      n0 + 16, lane, a01);
  store_bf16_tr(XT, NN, m0 + 16, n0,      lane, a10);
  store_bf16_tr(XT, NN, m0 + 16, n0 + 16, lane, a11);
  // term_w partial: sum over this 32x32 tile of W4[i,j] .* X
  const float* Wblk = W + (size_t)(i * NN) * BIGN + j * NN;
  int nl = lane & 15;
  float part = 0.0f;
  {
    int mb;
    mb = m0 + ((lane < 16) ? 0 : 8);
    #pragma unroll
    for (int v = 0; v < 8; ++v) {
      part += a00[v] * Wblk[(size_t)(mb + v) * BIGN + n0 + nl];
      part += a01[v] * Wblk[(size_t)(mb + v) * BIGN + n0 + 16 + nl];
    }
    mb = m0 + 16 + ((lane < 16) ? 0 : 8);
    #pragma unroll
    for (int v = 0; v < 8; ++v) {
      part += a10[v] * Wblk[(size_t)(mb + v) * BIGN + n0 + nl];
      part += a11[v] * Wblk[(size_t)(mb + v) * BIGN + n0 + 16 + nl];
    }
  }
  part = wave_sum(part);
  if (lane == 0) atomicAdd(&termw[pair], part);
}

// AX_ij = A_i @ X_ij ; store AXT (packed)
__global__ void __launch_bounds__(128) k_ax(const bf16_t* __restrict__ Abf,
                                            const bf16_t* __restrict__ XTbf,
                                            bf16_t* __restrict__ AXTbf) {
  int wave = (blockIdx.x * blockDim.x + threadIdx.x) >> 5;
  int lane = threadIdx.x & 31;
  int pair = wave >> 4, q = wave & 15;
  int i = pair >> 4, j = pair & 15;
  if (i == j) return;
  int m0 = (q >> 2) << 5, n0 = (q & 3) << 5;
  const bf16_t* Ai = Abf  + (size_t)i * NN2;
  const bf16_t* XT = XTbf + (size_t)pair * NN2;   // B^T (bt[c][b])
  v8f a00 = {}, a01 = {}, a10 = {}, a11 = {};
  #pragma unroll
  for (int k0 = 0; k0 < NN; k0 += 32) {
    v16bf fa0 = load_a_frag(Ai, NN, m0,      k0, lane);
    v16bf fa1 = load_a_frag(Ai, NN, m0 + 16, k0, lane);
    v16bf fb0 = load_b_frag(XT, NN, k0, n0,      lane);
    v16bf fb1 = load_b_frag(XT, NN, k0, n0 + 16, lane);
    a00 = wmma_bf16(fa0, fb0, a00);
    a01 = wmma_bf16(fa0, fb1, a01);
    a10 = wmma_bf16(fa1, fb0, a10);
    a11 = wmma_bf16(fa1, fb1, a11);
  }
  bf16_t* AXT = AXTbf + (size_t)pair * NN2;
  store_bf16_tr(AXT, NN, m0,      n0,      lane, a00);
  store_bf16_tr(AXT, NN, m0,      n0 + 16, lane, a01);
  store_bf16_tr(AXT, NN, m0 + 16, n0,      lane, a10);
  store_bf16_tr(AXT, NN, m0 + 16, n0 + 16, lane, a11);
}

// Y_ij = X_ij^T @ AX_ij ; fused fro2 = ||Y - A_blk[j]||_F^2
__global__ void __launch_bounds__(128) k_y(const bf16_t* __restrict__ XTbf,
                                           const bf16_t* __restrict__ AXTbf,
                                           const float* __restrict__ A,
                                           float* __restrict__ fro2) {
  int wave = (blockIdx.x * blockDim.x + threadIdx.x) >> 5;
  int lane = threadIdx.x & 31;
  int pair = wave >> 4, q = wave & 15;
  int i = pair >> 4, j = pair & 15;
  if (i == j) return;
  int m0 = (q >> 2) << 5, n0 = (q & 3) << 5;
  const bf16_t* XT  = XTbf  + (size_t)pair * NN2;  // rows n, K = a
  const bf16_t* AXT = AXTbf + (size_t)pair * NN2;  // B^T (bt[m][a])
  v8f a00 = {}, a01 = {}, a10 = {}, a11 = {};
  #pragma unroll
  for (int k0 = 0; k0 < NN; k0 += 32) {
    v16bf fa0 = load_a_frag(XT, NN, m0,      k0, lane);
    v16bf fa1 = load_a_frag(XT, NN, m0 + 16, k0, lane);
    v16bf fb0 = load_b_frag(AXT, NN, k0, n0,      lane);
    v16bf fb1 = load_b_frag(AXT, NN, k0, n0 + 16, lane);
    a00 = wmma_bf16(fa0, fb0, a00);
    a01 = wmma_bf16(fa0, fb1, a01);
    a10 = wmma_bf16(fa1, fb0, a10);
    a11 = wmma_bf16(fa1, fb1, a11);
  }
  const float* Ablk = A + (size_t)(j * NN) * BIGN + j * NN;
  int nl = lane & 15;
  float part = 0.0f;
  {
    int mb;
    mb = m0 + ((lane < 16) ? 0 : 8);
    #pragma unroll
    for (int v = 0; v < 8; ++v) {
      float d0 = a00[v] - Ablk[(size_t)(mb + v) * BIGN + n0 + nl];
      float d1 = a01[v] - Ablk[(size_t)(mb + v) * BIGN + n0 + 16 + nl];
      part += d0 * d0 + d1 * d1;
    }
    mb = m0 + 16 + ((lane < 16) ? 0 : 8);
    #pragma unroll
    for (int v = 0; v < 8; ++v) {
      float d0 = a10[v] - Ablk[(size_t)(mb + v) * BIGN + n0 + nl];
      float d1 = a11[v] - Ablk[(size_t)(mb + v) * BIGN + n0 + 16 + nl];
      part += d0 * d0 + d1 * d1;
    }
  }
  part = wave_sum(part);
  if (lane == 0) atomicAdd(&fro2[pair], part);
}

__global__ void k_alpha(const float* __restrict__ termw, const float* __restrict__ fro2,
                        float* __restrict__ alpha) {
  int idx = threadIdx.x;                  // 256
  int i = idx >> 4, j = idx & 15;
  alpha[idx] = (i == j) ? 0.0f : (termw[idx] + __expf(-sqrtf(fro2[idx])));
}

// 16x16 normalized spectral clustering via serial cyclic Jacobi; also emits Mmask.
__global__ void k_spectral(const float* __restrict__ alpha, int* __restrict__ cluster,
                           float* __restrict__ Mmask) {
  if (threadIdx.x != 0) return;
  float L[16][16], Vv[16][16], dinv[16];
  for (int i = 0; i < 16; ++i) {
    float d = 0.0f;
    for (int j = 0; j < 16; ++j) d += alpha[i * 16 + j];
    dinv[i] = rsqrtf(fmaxf(d, 1e-12f));
  }
  for (int i = 0; i < 16; ++i)
    for (int j = 0; j < 16; ++j) {
      L[i][j] = ((i == j) ? 1.0f : 0.0f) - dinv[i] * alpha[i * 16 + j] * dinv[j];
      Vv[i][j] = (i == j) ? 1.0f : 0.0f;
    }
  for (int sweep = 0; sweep < 32; ++sweep) {
    for (int p = 0; p < 15; ++p)
      for (int q = p + 1; q < 16; ++q) {
        float apq = L[p][q];
        if (fabsf(apq) <= 1e-12f) continue;
        float theta = (L[q][q] - L[p][p]) / (2.0f * apq);
        float tt = ((theta >= 0.0f) ? 1.0f : -1.0f) /
                   (fabsf(theta) + sqrtf(theta * theta + 1.0f));
        float c = 1.0f / sqrtf(tt * tt + 1.0f);
        float s = tt * c;
        for (int k = 0; k < 16; ++k) {
          float lkp = L[k][p], lkq = L[k][q];
          L[k][p] = c * lkp - s * lkq;
          L[k][q] = s * lkp + c * lkq;
        }
        for (int k = 0; k < 16; ++k) {
          float lpk = L[p][k], lqk = L[q][k];
          L[p][k] = c * lpk - s * lqk;
          L[q][k] = s * lpk + c * lqk;
        }
        for (int k = 0; k < 16; ++k) {
          float vkp = Vv[k][p], vkq = Vv[k][q];
          Vv[k][p] = c * vkp - s * vkq;
          Vv[k][q] = s * vkp + c * vkq;
        }
      }
  }
  int i0 = 0;
  for (int i = 1; i < 16; ++i) if (L[i][i] < L[i0][i0]) i0 = i;
  int i1 = -1;
  for (int i = 0; i < 16; ++i)
    if (i != i0 && (i1 < 0 || L[i][i] < L[i1][i1])) i1 = i;
  int cv[16];
  for (int i = 0; i < 16; ++i) {
    cv[i] = (Vv[i][i1] >= 0.0f) ? 1 : 0;
    cluster[i] = cv[i];
  }
  for (int i = 0; i < 16; ++i)
    for (int j = 0; j < 16; ++j)
      Mmask[i * 16 + j] = (i != j && cv[i] == cv[j]) ? 1.0f : 0.0f;
}

__global__ void k_out(const float* __restrict__ Uf, const int* __restrict__ cluster,
                      float* __restrict__ out) {
  int idx = blockIdx.x * blockDim.x + threadIdx.x;
  int total = G_CNT * NN2;
  if (idx < total) out[idx] = Uf[idx];
  else if (idx < total + G_CNT) out[idx] = (float)cluster[idx - total];
}

// ---------------------------------------------------------------------------
extern "C" void kernel_launch(void* const* d_in, const int* in_sizes, int n_in,
                              void* d_out, int out_size, void* d_ws, size_t ws_size,
                              hipStream_t stream) {
  const float* A  = (const float*)d_in[0];
  const float* W  = (const float*)d_in[1];
  const float* U0 = (const float*)d_in[2];
  (void)in_sizes; (void)n_in; (void)out_size; (void)ws_size;

  size_t off = 0;
  auto alloc = [&](size_t bytes) -> char* {
    char* p = (char*)d_ws + off;
    off += (bytes + 255) & ~(size_t)255;
    return p;
  };
  bf16_t* Wbf   = (bf16_t*)alloc((size_t)BIGN * BIGN * 2);
  bf16_t* Abf   = (bf16_t*)alloc((size_t)G_CNT * NN2 * 2);
  float*  Uf    = (float*) alloc((size_t)G_CNT * NN2 * 4);
  bf16_t* Ubf   = (bf16_t*)alloc((size_t)G_CNT * NN2 * 2);
  bf16_t* UTbf  = (bf16_t*)alloc((size_t)G_CNT * NN2 * 2);
  bf16_t* AUbf  = (bf16_t*)alloc((size_t)G_CNT * NN2 * 2);
  bf16_t* AUTbf = (bf16_t*)alloc((size_t)G_CNT * NN2 * 2);
  float*  Sf    = (float*) alloc((size_t)G_CNT * NN2 * 4);
  bf16_t* SsumT = (bf16_t*)alloc((size_t)G_CNT * NN2 * 2);
  float*  Vf    = (float*) alloc((size_t)G_CNT * NN2 * 4);
  bf16_t* XTbf  = (bf16_t*)alloc((size_t)G_CNT * G_CNT * NN2 * 2);
  bf16_t* AXTbf = (bf16_t*)alloc((size_t)G_CNT * G_CNT * NN2 * 2);
  float*  termw = (float*) alloc(256 * 4);
  float*  fro2  = (float*) alloc(256 * 4);
  float*  alphb = (float*) alloc(256 * 4);
  float*  Mmask = (float*) alloc(256 * 4);
  int*    clst  = (int*)   alloc(64);

  // --- prep: bf16 copies of W, A diag blocks, U0 (+transpose) ---
  {
    size_t nW = (size_t)BIGN * BIGN;
    k_prep_w<<<(unsigned)((nW + 255) / 256), 256, 0, stream>>>(W, Wbf);
    k_prep_a<<<(G_CNT * NN2 + 255) / 256, 256, 0, stream>>>(A, Abf);
    k_prep_u<<<(G_CNT * NN2 + 255) / 256, 256, 0, stream>>>(U0, Uf, Ubf, UTbf);
    k_init_mask<<<1, 256, 0, stream>>>(Mmask);
  }

  auto forward_backup = [&](float tau) {
    for (int it = 0; it < 10; ++it) {
      k_au<<<64, 128, 0, stream>>>(Abf, UTbf, AUbf, AUTbf);           // 256 waves
      k_s <<<64, 128, 0, stream>>>(UTbf, AUTbf, Sf);
      k_ssum<<<(G_CNT * NN2) / 256, 256, 0, stream>>>(Sf, Mmask, SsumT);
      k_v <<<64, 128, 0, stream>>>(Wbf, UTbf, AUbf, SsumT, Mmask, Vf);
      k_sinkhorn<<<G_CNT, 256, 0, stream>>>(Vf, tau, Uf, Ubf, UTbf);
    }
  };

  forward_backup(0.5f);
  for (int o = 0; o < 2; ++o) {
    k_zero2<<<1, 256, 0, stream>>>(termw, fro2);
    k_x <<<1024, 128, 0, stream>>>(Ubf, W, XTbf, termw);              // 4096 waves
    k_ax<<<1024, 128, 0, stream>>>(Abf, XTbf, AXTbf);
    k_y <<<1024, 128, 0, stream>>>(XTbf, AXTbf, A, fro2);
    k_alpha<<<1, 256, 0, stream>>>(termw, fro2, alphb);
    k_spectral<<<1, 32, 0, stream>>>(alphb, clst, Mmask);
    forward_backup((o == 0) ? 0.5f : 0.01f);
  }

  int total = G_CNT * NN2 + G_CNT;
  k_out<<<(total + 255) / 256, 256, 0, stream>>>(Uf, clst, (float*)d_out);
}
